// CGNN_46840913330460
// MI455X (gfx1250) — matrix operand order repaired and checked by reading, not compiled, not run on hardware
//
#include <hip/hip_runtime.h>
#include <hip/hip_bf16.h>

// ---------------------------------------------------------------------------
// Sizes (compile-time constants matching the reference)
// ---------------------------------------------------------------------------
#define NB      64          // batch
#define CIN     768
#define PIX     1024        // 32*32
#define NNODES  65536       // NB*PIX
#define HID     128
#define NEDGE   524288
#define NCLS    1000
#define BN_EPS  1e-5f

typedef __bf16 bf16;
typedef __attribute__((ext_vector_type(16))) __bf16 v16bf;
typedef __attribute__((ext_vector_type(8)))  __bf16 v8bf;
typedef __attribute__((ext_vector_type(8)))  float  v8f;

// ---------------------------------------------------------------------------
// CDNA5 async-copy helpers (ASYNCcnt-tracked DMA: global -> LDS)
// ---------------------------------------------------------------------------
__device__ __forceinline__ void async_b128_to_lds(unsigned lds_off, const void* gptr) {
    // per-lane: LDS[lds_off] <= MEM[gptr] (16 bytes), tracked with ASYNCcnt
    asm volatile("global_load_async_to_lds_b128 %0, %1, off"
                 :: "v"(lds_off), "v"((unsigned long long)(uintptr_t)gptr)
                 : "memory");
}
#define WAIT_ASYNCCNT(N) asm volatile("s_wait_asynccnt %0" :: "n"(N) : "memory")

// Stage one 32(K) x 128(O) bf16 B-slice (8KB) into LDS, cooperatively.
// 512 x 16B chunks; each of 256 threads issues 2 async copies.
__device__ __forceinline__ void stage_b_slice(const bf16* __restrict__ Wbf, int kdim,
                                              unsigned sbase, int buf, int c0, int tid) {
    #pragma unroll
    for (int j = 0; j < 2; ++j) {
        const int c    = tid + (j << 8);   // chunk id 0..511
        const int o    = c >> 2;           // output channel (row of W)
        const int part = c & 3;            // 16B chunk within the 64B k-slice row
        const unsigned loff = sbase + (unsigned)(buf * (32 * HID * 2) + o * 64 + part * 16);
        async_b128_to_lds(loff, Wbf + (size_t)o * kdim + c0 + part * 8);
    }
}

// ---------------------------------------------------------------------------
// WMMA GEMM:  out[M][128] = A[M][KDIM] * W[128][KDIM]^T  (+ bias)
//  - A either comes from x ([B][CIN][PIX] fp32, node n = b*1024+p -> A[n][c] =
//    x[b][c][p], converted to bf16 in-register) or from a row-major bf16 array.
//  - Each wave owns one 16-row tile and ALL 8 column tiles (8 v8f accums) so
//    the A stream (the 201 MB x tensor) is read exactly once from HBM.
//  - B k-slices are staged in LDS (double-buffered async DMA) and shared by
//    all 8 waves of the block: L2 B-traffic drops 8x vs per-wave global loads.
// ---------------------------------------------------------------------------
template<int KDIM, bool A_FROM_X>
__global__ __launch_bounds__(256)
void wmma_gemm_kernel(const float* __restrict__ X,
                      const bf16*  __restrict__ Abf,
                      const bf16*  __restrict__ Wbf,   // [128][KDIM] row-major bf16
                      const float* __restrict__ bias,  // [128] or nullptr
                      float* __restrict__ out)         // [M][128]
{
    __shared__ __align__(64) bf16 sB[2][32 * HID];     // 2 x 8KB double buffer

    const int tid  = threadIdx.x;
    const int wave = tid >> 5;
    const int lane = tid & 31;
    const int m    = lane & 15;      // row within tile (A) / column (B,D)
    const int half = lane >> 4;      // lane group 0/1
    const int rowTile = blockIdx.x * 8 + wave;
    const int n0 = rowTile << 4;     // first of 16 rows

    const unsigned sbase = (unsigned)(uintptr_t)(void*)&sB[0][0];

    v8f acc[8];
    #pragma unroll
    for (int t = 0; t < 8; ++t) acc[t] = (v8f){0.f,0.f,0.f,0.f,0.f,0.f,0.f,0.f};

    constexpr int STEPS = KDIM / 32;
    stage_b_slice(Wbf, KDIM, sbase, 0, 0, tid);        // prologue: slice 0

    for (int ks = 0; ks < STEPS; ++ks) {
        const int c0 = ks * 32;
        if (ks + 1 < STEPS) {
            stage_b_slice(Wbf, KDIM, sbase, (ks + 1) & 1, c0 + 32, tid);
            WAIT_ASYNCCNT(2);        // own slice-ks copies retired (in-order)
        } else {
            WAIT_ASYNCCNT(0);
        }
        __syncthreads();             // slice ks fully visible in LDS to all waves

        // ---- A fragment: K(i) = i + 16*half (contiguous, matches B from LDS)
        v16bf a;
        if (A_FROM_X) {
            const int b = n0 >> 10;
            const int p = (n0 & 1023) + m;
            const float* xa = X + ((size_t)b * KDIM) * PIX + p;
            #pragma unroll
            for (int i = 0; i < 16; ++i) {
                const int k = c0 + i + 16 * half;
                a[i] = (bf16)xa[(size_t)k * PIX];   // coalesced across lanes 0..15
            }
        } else {
            const bf16* ar = Abf + (size_t)(n0 + m) * KDIM + c0 + 16 * half;
            v8bf lo = *(const v8bf*)(ar);        // K = c0+16h .. +7
            v8bf hi = *(const v8bf*)(ar + 8);    // K = c0+16h+8 .. +7
            #pragma unroll
            for (int i = 0; i < 8; ++i) { a[i] = lo[i]; a[i + 8] = hi[i]; }
        }

        // ---- 8 column tiles share this A fragment; B fragments from LDS
        const bf16* sBcur = &sB[ks & 1][0];
        #pragma unroll
        for (int t = 0; t < 8; ++t) {
            const int o = (t << 4) + m;          // output channel = B column
            v16bf bfrag = *(const v16bf*)(sBcur + o * 32 + half * 16); // K = c0+16h..+15
            acc[t] = __builtin_amdgcn_wmma_f32_16x16x32_bf16(
                false, a, false, bfrag, (short)0, acc[t], false, false);
        }
        __syncthreads();             // all waves done reading before buffer reuse
    }

    // ---- store: lane col = m, rows r + 8*half (ISA 32-bit 16x16 C/D layout)
    #pragma unroll
    for (int t = 0; t < 8; ++t) {
        const int o  = (t << 4) + m;
        const float bv = (bias != nullptr) ? bias[o] : 0.0f;
        #pragma unroll
        for (int r = 0; r < 8; ++r) {
            const int row = n0 + r + 8 * half;
            out[(size_t)row * HID + o] = acc[t][r] + bv;
        }
    }
}

// ---------------------------------------------------------------------------
// Small helpers
// ---------------------------------------------------------------------------
__global__ void zero_f32_kernel(float* __restrict__ p, int n) {
    int i = blockIdx.x * blockDim.x + threadIdx.x;
    if (i < n) p[i] = 0.0f;
}

__global__ void f32_to_bf16_kernel(const float* __restrict__ in,
                                   bf16* __restrict__ out, int n) {
    int i = blockIdx.x * blockDim.x + threadIdx.x;
    if (i < n) out[i] = (bf16)in[i];
}

// Per-channel sum / sumsq over rows ([rows][128]); LDS ds_add_f32 reduction.
__global__ __launch_bounds__(256)
void col_stats_kernel(const float* __restrict__ y, float* __restrict__ sums,
                      int rowsPerBlock) {
    __shared__ float ssum[HID];
    __shared__ float ssq[HID];
    if (threadIdx.x < HID) { ssum[threadIdx.x] = 0.f; ssq[threadIdx.x] = 0.f; }
    __syncthreads();
    const int c   = threadIdx.x & 127;
    const int sub = threadIdx.x >> 7;           // 0..1
    const int r0  = blockIdx.x * rowsPerBlock;
    float ls = 0.f, lq = 0.f;
    for (int r = sub; r < rowsPerBlock; r += 2) {
        float v = y[(size_t)(r0 + r) * HID + c];
        ls += v; lq += v * v;
    }
    atomicAdd(&ssum[c], ls);
    atomicAdd(&ssq[c], lq);
    __syncthreads();
    if (threadIdx.x < HID) {
        atomicAdd(&sums[threadIdx.x],       ssum[threadIdx.x]);
        atomicAdd(&sums[HID + threadIdx.x], ssq[threadIdx.x]);
    }
}

// sums[0:128)=sum, [128:256)=sumsq  ->  ss[0:128)=scale, [128:256)=shift
__global__ void bn_finalize_kernel(const float* __restrict__ sums,
                                   const float* __restrict__ g,
                                   const float* __restrict__ b,
                                   float* __restrict__ ss, float invN) {
    int o = threadIdx.x;
    if (o < HID) {
        float mu  = sums[o] * invN;
        float var = sums[HID + o] * invN - mu * mu;
        float s   = g[o] * rsqrtf(var + BN_EPS);
        ss[o]       = s;
        ss[HID + o] = b[o] - mu * s;
    }
}

// y -> relu(y*scale+shift) as bf16 nodes
__global__ __launch_bounds__(256)
void bn_relu_bf16_kernel(const float* __restrict__ y,
                         const float* __restrict__ ss,
                         bf16* __restrict__ out, int total) {
    int i = blockIdx.x * blockDim.x + threadIdx.x;
    if (i < total) {
        int c = i & 127;
        float v = fmaxf(y[i] * ss[c] + ss[HID + c], 0.f);
        out[i] = (bf16)v;
    }
}

__global__ void degree_kernel(const int* __restrict__ dst,
                              float* __restrict__ degf, int E) {
    int e = blockIdx.x * blockDim.x + threadIdx.x;
    if (e < E) atomicAdd(&degf[dst[e]], 1.0f);
}

__global__ void dis_kernel(const float* __restrict__ degf,
                           float* __restrict__ dis, int n) {
    int i = blockIdx.x * blockDim.x + threadIdx.x;
    if (i < n) dis[i] = rsqrtf(degf[i] + 1.0f);   // +1 self-loop, always > 0
}

// h2 = h1 * dis^2 + gcn_b   (self-loop term, also initializes the accumulator)
__global__ __launch_bounds__(256)
void selfloop_init_kernel(const float* __restrict__ h1,
                          const float* __restrict__ dis,
                          const float* __restrict__ gcn_b,
                          float* __restrict__ h2, int total) {
    int i = blockIdx.x * blockDim.x + threadIdx.x;
    if (i < total) {
        int n = i >> 7, o = i & 127;
        float d = dis[n];
        h2[i] = h1[i] * d * d + gcn_b[o];
    }
}

// One wave per edge; each lane handles 4 channels (float4 load + 4 atomics).
__global__ __launch_bounds__(256)
void scatter_kernel(const int* __restrict__ src, const int* __restrict__ dst,
                    const float* __restrict__ h1, const float* __restrict__ dis,
                    float* __restrict__ h2, int E) {
    int gid  = blockIdx.x * blockDim.x + threadIdx.x;
    int e    = gid >> 5;
    int lane = gid & 31;
    if (e >= E) return;
    int s = src[e], d = dst[e];
    float norm = dis[s] * dis[d];
    const float4 v = ((const float4*)(h1 + (size_t)s * HID))[lane];
    float* hd = h2 + (size_t)d * HID + lane * 4;
    atomicAdd(hd + 0, v.x * norm);
    atomicAdd(hd + 1, v.y * norm);
    atomicAdd(hd + 2, v.z * norm);
    atomicAdd(hd + 3, v.w * norm);
}

// In-place BN1d + ReLU on h, and gate[n] = dot(h[n,:], gate_w) + gate_b
__global__ __launch_bounds__(256)
void bn_relu_gate_kernel(float* __restrict__ h, const float* __restrict__ ss,
                         const float* __restrict__ gate_w,
                         const float* __restrict__ gate_b,
                         float* __restrict__ gate) {
    __shared__ float red[256];
    const int sub = threadIdx.x >> 7;            // row within block (0/1)
    const int o   = threadIdx.x & 127;
    const int n   = blockIdx.x * 2 + sub;
    const size_t idx = (size_t)n * HID + o;
    float v = fmaxf(h[idx] * ss[o] + ss[HID + o], 0.f);
    h[idx] = v;
    red[threadIdx.x] = v * gate_w[o];
    __syncthreads();
    for (int s = 64; s > 0; s >>= 1) {
        if (o < s) red[threadIdx.x] += red[threadIdx.x + s];
        __syncthreads();
    }
    if (o == 0) gate[n] = red[sub * 128] + gate_b[0];
}

// One block per graph (contiguous 1024 nodes): segment softmax + att-pooling.
__global__ __launch_bounds__(256)
void pool_kernel(const float* __restrict__ hfeat, const float* __restrict__ gate,
                 float* __restrict__ att_out, float* __restrict__ pooled_out) {
    __shared__ float red[256];
    __shared__ float s_max, s_den;
    const int g = blockIdx.x;
    const int t = threadIdx.x;
    const float* gg = gate + (size_t)g * PIX;

    float mx = -3.4e38f;
    for (int i = t; i < PIX; i += 256) mx = fmaxf(mx, gg[i]);
    red[t] = mx; __syncthreads();
    for (int s = 128; s > 0; s >>= 1) { if (t < s) red[t] = fmaxf(red[t], red[t + s]); __syncthreads(); }
    if (t == 0) s_max = red[0];
    __syncthreads();

    float se = 0.f;
    for (int i = t; i < PIX; i += 256) se += __expf(gg[i] - s_max);
    red[t] = se; __syncthreads();
    for (int s = 128; s > 0; s >>= 1) { if (t < s) red[t] += red[t + s]; __syncthreads(); }
    if (t == 0) s_den = red[0];
    __syncthreads();

    const float inv = 1.0f / s_den;
    for (int i = t; i < PIX; i += 256)
        att_out[(size_t)g * PIX + i] = __expf(gg[i] - s_max) * inv;

    const int o = t & 127, sub = t >> 7;
    float acc = 0.f;
    for (int r = sub; r < PIX; r += 2) {
        float a = __expf(gg[r] - s_max) * inv;
        acc += a * hfeat[((size_t)g * PIX + r) * HID + o];   // relu(h)==h here
    }
    red[t] = acc; __syncthreads();
    if (t < 128) pooled_out[(size_t)g * HID + t] = red[t] + red[t + 128];
}

// logits[g][cls] = pooled[g,:] . fc_w[cls,:] + fc_b[cls]
__global__ __launch_bounds__(256)
void fc_kernel(const float* __restrict__ pooled, const float* __restrict__ fc_w,
               const float* __restrict__ fc_b, float* __restrict__ logits) {
    int i = blockIdx.x * blockDim.x + threadIdx.x;
    if (i >= NB * NCLS) return;
    int g = i / NCLS, cls = i - g * NCLS;
    const float* p = pooled + (size_t)g * HID;
    const float* w = fc_w + (size_t)cls * HID;
    float acc = fc_b[cls];
    #pragma unroll 8
    for (int k = 0; k < HID; ++k) acc += p[k] * w[k];
    logits[i] = acc;
}

// ---------------------------------------------------------------------------
// Launch
// ---------------------------------------------------------------------------
extern "C" void kernel_launch(void* const* d_in, const int* in_sizes, int n_in,
                              void* d_out, int out_size, void* d_ws, size_t ws_size,
                              hipStream_t stream) {
    const float* x       = (const float*)d_in[0];
    const int*   eidx    = (const int*)  d_in[1];   // [2][E]
    const float* conv_w  = (const float*)d_in[2];
    const float* conv_b  = (const float*)d_in[3];
    const float* bn2_g   = (const float*)d_in[4];
    const float* bn2_b   = (const float*)d_in[5];
    const float* gcn_w   = (const float*)d_in[6];
    const float* gcn_b   = (const float*)d_in[7];
    const float* bn1_g   = (const float*)d_in[8];
    const float* bn1_b   = (const float*)d_in[9];
    const float* gate_w  = (const float*)d_in[10];
    const float* gate_b  = (const float*)d_in[11];
    const float* fc_w    = (const float*)d_in[12];
    const float* fc_b    = (const float*)d_in[13];

    const int* esrc = eidx;
    const int* edst = eidx + NEDGE;

    // workspace carving (bytes)
    char* ws = (char*)d_ws;
    float* y_h2   = (float*)(ws);                                   // N*128 f32 (y, reused as h2)
    float* h1     = (float*)(ws + (size_t)NNODES * HID * 4);        // N*128 f32
    bf16*  nodes  = (bf16*) (ws + (size_t)NNODES * HID * 8);        // N*128 bf16
    bf16*  convwb = (bf16*) (ws + (size_t)NNODES * HID * 10);       // 128*768 bf16
    bf16*  gcnwb  = (bf16*) (ws + (size_t)NNODES * HID * 10 + 196608);
    float* degf   = (float*)(ws + (size_t)NNODES * HID * 10 + 196608 + 32768);
    float* dis    = degf + NNODES;
    float* gate   = dis  + NNODES;
    float* stats  = gate + NNODES;          // [0:256) sums, [256:512) scale/shift

    float* out_logits = (float*)d_out;                  // 64*1000
    float* out_att    = out_logits + NB * NCLS;         // 65536
    float* out_pooled = out_att + NNODES;               // 64*128

    const int T = 256;
    const int totNH = NNODES * HID;

    // 0) per-call init (graph-replay safe)
    zero_f32_kernel<<<NNODES / T, T, 0, stream>>>(degf, NNODES);
    zero_f32_kernel<<<1, T, 0, stream>>>(stats, 256);
    f32_to_bf16_kernel<<<(HID * CIN) / T, T, 0, stream>>>(conv_w, convwb, HID * CIN);
    f32_to_bf16_kernel<<<(HID * HID) / T, T, 0, stream>>>(gcn_w, gcnwb, HID * HID);

    // 1) conv_down GEMM (x streamed once from HBM, bf16 WMMA, async B staging)
    wmma_gemm_kernel<CIN, true><<<NNODES / 16 / 8, T, 0, stream>>>(
        x, nullptr, convwb, conv_b, y_h2);

    // 2) BN2d (training-mode stats) + ReLU -> bf16 nodes
    col_stats_kernel<<<256, T, 0, stream>>>(y_h2, stats, NNODES / 256);
    bn_finalize_kernel<<<1, 128, 0, stream>>>(stats, bn2_g, bn2_b, stats + 256,
                                              1.0f / (float)NNODES);
    bn_relu_bf16_kernel<<<totNH / T, T, 0, stream>>>(y_h2, stats + 256, nodes, totNH);

    // 3) GCN weight GEMM: h1 = nodes @ gcn_w^T
    wmma_gemm_kernel<HID, false><<<NNODES / 16 / 8, T, 0, stream>>>(
        nullptr, nodes, gcnwb, nullptr, h1);

    // 4) symmetric-norm aggregation (deg -> dis -> self-loop init -> edge scatter)
    degree_kernel<<<NEDGE / T, T, 0, stream>>>(edst, degf, NEDGE);
    dis_kernel<<<NNODES / T, T, 0, stream>>>(degf, dis, NNODES);
    selfloop_init_kernel<<<totNH / T, T, 0, stream>>>(h1, dis, gcn_b, y_h2, totNH);
    scatter_kernel<<<(NEDGE * 32) / T, T, 0, stream>>>(esrc, edst, h1, dis, y_h2, NEDGE);

    // 5) BN1d + ReLU (in place) + gate projection
    zero_f32_kernel<<<1, T, 0, stream>>>(stats, 256);
    col_stats_kernel<<<256, T, 0, stream>>>(y_h2, stats, NNODES / 256);
    bn_finalize_kernel<<<1, 128, 0, stream>>>(stats, bn1_g, bn1_b, stats + 256,
                                              1.0f / (float)NNODES);
    bn_relu_gate_kernel<<<NNODES / 2, T, 0, stream>>>(y_h2, stats + 256,
                                                      gate_w, gate_b, gate);

    // 6) per-graph softmax attention pooling (writes att + pooled outputs)
    pool_kernel<<<NB, T, 0, stream>>>(y_h2, gate, out_att, out_pooled);

    // 7) classifier head
    fc_kernel<<<(NB * NCLS + T - 1) / T, T, 0, stream>>>(out_pooled, fc_w, fc_b,
                                                         out_logits);
}
